// DeepSets_57535381897776
// MI455X (gfx1250) — compile-verified, matrix operand order checked by reading, not run on hardware
//
#include <hip/hip_runtime.h>

// MI455X / gfx1250: wave32, WMMA f32_16x16x32_f16 + Tensor Data Mover path.
typedef __attribute__((ext_vector_type(16))) _Float16 v16h;
typedef __attribute__((ext_vector_type(8)))  float    v8f;
typedef __attribute__((ext_vector_type(4)))  unsigned v4u;
typedef __attribute__((ext_vector_type(8)))  int      v8i;
typedef __attribute__((ext_vector_type(4)))  int      v4i;

#define EPSV      1e-5f
#define NSEG      2048
#define C_IN      128
#define HDIM      128
#define ODIM      64
#define AGGD      (3 * HDIM)
#define TILE_ROWS 64
#define THREADS   128   // 4 waves of 32

union AFrag { v16h v; _Float16 e[16]; };
union DFrag { v8f  v; float    e[8];  };

// ---------------------------------------------------------------------------
// TDM: issue one 2-D tensor_load_to_lds (f16 elements, contiguous rows).
// D# bit layout per CDNA5 ISA ch.8: group0 = {count, lds_addr, global_addr,
// type=2}; group1 = {data_size, tensor_dim0/1, tile_dim0/1, dim0 stride}.
// Groups 2/3 zero (<=2-D tile). 6-arg builtin form (clang-23 lane).
// Tracked by TENSORcnt.
// ---------------------------------------------------------------------------
__device__ __forceinline__ void tdm_load_2d_f16(unsigned lds_off, const void* gptr,
                                                int rows, int cols, int stride_elems)
{
    const unsigned long long ga = (unsigned long long)gptr;
    v4u g0;
    g0[0] = 1u;                                    // count=1, user descriptor
    g0[1] = lds_off;                               // lds_addr (bytes)
    g0[2] = (unsigned)(ga & 0xffffffffu);          // global_addr[31:0]
    g0[3] = (unsigned)((ga >> 32) & 0x1ffffffu)    // global_addr[56:32]
          | (2u << 30);                            // type=2 ("image")
    v8i g1;
    g1[0] = (1 << 16);                             // wg_mask=0, data_size=1 (2B)
    g1[1] = (cols & 0xffff) << 16;                 // tensor_dim0[15:0]
    g1[2] = ((cols >> 16) & 0xffff)                // tensor_dim0[31:16]
          | ((rows & 0xffff) << 16);               // tensor_dim1[15:0]
    g1[3] = ((rows >> 16) & 0xffff)                // tensor_dim1[31:16]
          | ((cols & 0xffff) << 16);               // tile_dim0
    g1[4] = (rows & 0xffff);                       // tile_dim1 (tile_dim2=0)
    g1[5] = stride_elems;                          // tensor_dim0_stride[31:0]
    g1[6] = 0;                                     // stride hi / dim1_stride lo
    g1[7] = 0;
    const v4i gz4 = {0, 0, 0, 0};
    const v8i gz8 = {0, 0, 0, 0, 0, 0, 0, 0};
    __builtin_amdgcn_tensor_load_to_lds(g0, g1, gz4, gz4, gz8, 0);
}

// ---------------------------------------------------------------------------
// Kernel 1: zero segment accumulators, convert weights to f16, fold BN.
//   relu(bn(xW^T + b)) == relu((xW^T) * s + t),
//   s = g*rsqrt(v+eps), t = b*s + be - m*s
// ---------------------------------------------------------------------------
__global__ __launch_bounds__(256) void prep_kernel(
    const float* __restrict__ W1, const float* __restrict__ W2, const float* __restrict__ W3,
    const float* __restrict__ b1, const float* __restrict__ g1, const float* __restrict__ be1,
    const float* __restrict__ m1, const float* __restrict__ v1,
    const float* __restrict__ b2, const float* __restrict__ g2, const float* __restrict__ be2,
    const float* __restrict__ m2, const float* __restrict__ v2,
    const float* __restrict__ b3, const float* __restrict__ g3, const float* __restrict__ be3,
    const float* __restrict__ m3, const float* __restrict__ v3,
    _Float16* __restrict__ Wh, float* __restrict__ sc, float* __restrict__ tb,
    float* __restrict__ gsum, unsigned* __restrict__ gmax, int* __restrict__ gcnt)
{
    const int idx = blockIdx.x * blockDim.x + threadIdx.x;
    if (idx < NSEG * HDIM) { gsum[idx] = 0.0f; gmax[idx] = 0u; }
    if (idx < NSEG) gcnt[idx] = 0;
    if (idx < 3 * HDIM * C_IN) {
        const int l = idx / (HDIM * C_IN);
        const int rem = idx % (HDIM * C_IN);
        const float* W = (l == 0) ? W1 : (l == 1) ? W2 : W3;
        Wh[idx] = (_Float16)W[rem];
    }
    if (idx < 3 * HDIM) {
        const int l = idx / HDIM, j = idx % HDIM;
        const float* b  = (l == 0) ? b1  : (l == 1) ? b2  : b3;
        const float* g  = (l == 0) ? g1  : (l == 1) ? g2  : g3;
        const float* be = (l == 0) ? be1 : (l == 1) ? be2 : be3;
        const float* m  = (l == 0) ? m1  : (l == 1) ? m2  : m3;
        const float* v  = (l == 0) ? v1  : (l == 1) ? v2  : v3;
        const float s = g[j] * rsqrtf(v[j] + EPSV);
        sc[idx] = s;
        tb[idx] = b[j] * s + be[j] - m[j] * s;
    }
}

// ---------------------------------------------------------------------------
// Kernel 2: fused 3-layer MLP (WMMA f16->f32, TDM weight staging) + local
// segmented sum/max/count with per-segment global atomics (batch is sorted).
// ---------------------------------------------------------------------------
__global__ __launch_bounds__(THREADS) void mlp_agg_kernel(
    const float* __restrict__ x, const int* __restrict__ batch,
    const _Float16* __restrict__ Wh,
    const float* __restrict__ sc, const float* __restrict__ tb,
    float* __restrict__ gsum, unsigned* __restrict__ gmax, int* __restrict__ gcnt,
    int N)
{
    __shared__ _Float16 act[TILE_ROWS * HDIM];   // 16 KB activation tile (in-place)
    __shared__ _Float16 wlds[HDIM * C_IN];       // 32 KB current-layer weights (TDM dest)
    __shared__ float    ssc[3 * HDIM];           // folded BN scales
    __shared__ float    stb[3 * HDIM];           // folded BN biases
    __shared__ int      sbatch[TILE_ROWS];

    const int tid  = threadIdx.x;
    const int lane = tid & 31;
    const int wid  = tid >> 5;           // wave id 0..3, owns rows [wid*16, wid*16+16)
    const int r    = lane & 15;
    const int hs   = lane >> 4;          // lane half-select
    const int r0   = blockIdx.x * TILE_ROWS;

    // --- load x tile as f16 (zero-pad past N), batch ids, BN scale/bias ---
    for (int idx = tid; idx < TILE_ROWS * C_IN; idx += THREADS) {
        const int row = idx >> 7, col = idx & 127;
        const long grow = (long)r0 + row;
        const float val = (grow < (long)N) ? x[grow * C_IN + col] : 0.0f;
        act[idx] = (_Float16)val;
    }
    if (tid < TILE_ROWS) {
        const int grow = r0 + tid;
        sbatch[tid] = (grow < N) ? batch[grow] : -1;
    }
    for (int idx = tid; idx < 3 * HDIM; idx += THREADS) {
        ssc[idx] = sc[idx];
        stb[idx] = tb[idx];
    }
    __syncthreads();

    // per-lane K offsets for the 16-bit 16x32 A/B fragment layout
    int kofs[8];
#pragma unroll
    for (int i = 0; i < 8; ++i)
        kofs[i] = ((i < 4) ? 2 * i : 16 + 2 * (i - 4)) + hs * 8;

    const unsigned wlds_off = (unsigned)(size_t)&wlds[0];

    for (int layer = 0; layer < 3; ++layer) {
        __syncthreads();  // everyone done reading previous wlds
        // async DMA this layer's 128x128 f16 weights into LDS (wave 0 only)
        if (wid == 0) {
            tdm_load_2d_f16(wlds_off, Wh + layer * HDIM * C_IN,
                            HDIM, C_IN, C_IN);
        }

        // overlap DMA with A-fragment loads from act (own rows, untouched by TDM)
        AFrag a[4];
        const int arow = wid * 16 + r;
#pragma unroll
        for (int kc = 0; kc < 4; ++kc) {
            const _Float16* ap = &act[arow * HDIM + kc * 32];
#pragma unroll
            for (int i = 0; i < 8; ++i) {
                a[kc].e[2 * i]     = ap[kofs[i]];
                a[kc].e[2 * i + 1] = ap[kofs[i] + 1];
            }
        }

        if (wid == 0) __builtin_amdgcn_s_wait_tensorcnt(0);
        __syncthreads();

        // sweep the 8 output-channel tiles of this wave's 16-row strip
#pragma unroll
        for (int nt = 0; nt < 8; ++nt) {
            DFrag dacc;
#pragma unroll
            for (int i = 0; i < 8; ++i) dacc.e[i] = 0.0f;

#pragma unroll
            for (int kc = 0; kc < 4; ++kc) {
                AFrag b;   // B = W^T : lane's column n == weight row nout
                const _Float16* bp = &wlds[(nt * 16 + r) * C_IN + kc * 32];
#pragma unroll
                for (int i = 0; i < 8; ++i) {
                    b.e[2 * i]     = bp[kofs[i]];
                    b.e[2 * i + 1] = bp[kofs[i] + 1];
                }
                dacc.v = __builtin_amdgcn_wmma_f32_16x16x32_f16(
                    false, a[kc].v, false, b.v, (short)0, dacc.v, false, false);
            }

            // fused BN + ReLU epilogue from LDS consts, f16 write-back in place
            const int nout = nt * 16 + r;
            const float s = ssc[layer * HDIM + nout];
            const float t = stb[layer * HDIM + nout];
#pragma unroll
            for (int i = 0; i < 8; ++i) {
                const int row = wid * 16 + hs * 8 + i;
                const float h = fmaxf(dacc.e[i] * s + t, 0.0f);
                act[row * HDIM + nout] = (_Float16)h;
            }
        }
    }
    __syncthreads();

    // --- segmented sum/max/count; batch sorted -> tile spans few segments ---
    const int nvalid = min(TILE_ROWS, N - r0);
    const int b0 = sbatch[0];
    const int b1 = sbatch[nvalid - 1];
    const int c = tid;  // THREADS == HDIM: one channel per thread
    for (int sid = b0; sid <= b1; ++sid) {
        float s = 0.0f, mx = 0.0f;
        int hit = 0;
        for (int lr = 0; lr < nvalid; ++lr) {
            if (sbatch[lr] == sid) {
                const float v = (float)act[lr * HDIM + c];
                s += v;
                mx = fmaxf(mx, v);
                ++hit;
            }
        }
        if (hit) {
            atomicAdd(&gsum[sid * HDIM + c], s);
            // post-ReLU values are >= 0: float bits compare monotonically as uint
            atomicMax(&gmax[sid * HDIM + c], __float_as_uint(mx));
            if (c == 0) atomicAdd(&gcnt[sid], hit);
        }
    }
}

// ---------------------------------------------------------------------------
// Kernel 3: per-segment concat [sum|max|mean], tiny 384->64 GEMM + BN (f32).
// ---------------------------------------------------------------------------
__global__ __launch_bounds__(ODIM) void out_kernel(
    const float* __restrict__ gsum, const unsigned* __restrict__ gmax,
    const int* __restrict__ gcnt,
    const float* __restrict__ Wo, const float* __restrict__ bo,
    const float* __restrict__ go, const float* __restrict__ beo,
    const float* __restrict__ mo, const float* __restrict__ vo,
    float* __restrict__ out)
{
    __shared__ float agg[AGGD];
    const int s = blockIdx.x;
    const int tid = threadIdx.x;
    const float cnt = (float)gcnt[s];
    const float inv = 1.0f / fmaxf(cnt, 1.0f);
    for (int j = tid; j < HDIM; j += ODIM) {
        const float sm = gsum[s * HDIM + j];
        agg[j] = sm;
        agg[HDIM + j] = __uint_as_float(gmax[s * HDIM + j]); // 0 for empty segs
        agg[2 * HDIM + j] = sm * inv;
    }
    __syncthreads();
    const int o = tid;
    float acc = bo[o];
    for (int j = 0; j < AGGD; ++j)
        acc = fmaf(agg[j], Wo[o * AGGD + j], acc);
    const float sco = go[o] * rsqrtf(vo[o] + EPSV);
    out[s * ODIM + o] = (acc - mo[o]) * sco + beo[o];
}

// ---------------------------------------------------------------------------
extern "C" void kernel_launch(void* const* d_in, const int* in_sizes, int n_in,
                              void* d_out, int out_size, void* d_ws, size_t ws_size,
                              hipStream_t stream)
{
    const float* x     = (const float*)d_in[0];
    const int*   batch = (const int*)  d_in[1];
    const float* W1 = (const float*)d_in[2],  *b1 = (const float*)d_in[3];
    const float* g1 = (const float*)d_in[4],  *be1 = (const float*)d_in[5];
    const float* m1 = (const float*)d_in[6],  *v1 = (const float*)d_in[7];
    const float* W2 = (const float*)d_in[8],  *b2 = (const float*)d_in[9];
    const float* g2 = (const float*)d_in[10], *be2 = (const float*)d_in[11];
    const float* m2 = (const float*)d_in[12], *v2 = (const float*)d_in[13];
    const float* W3 = (const float*)d_in[14], *b3 = (const float*)d_in[15];
    const float* g3 = (const float*)d_in[16], *be3 = (const float*)d_in[17];
    const float* m3 = (const float*)d_in[18], *v3 = (const float*)d_in[19];
    const float* Wo = (const float*)d_in[20], *bo = (const float*)d_in[21];
    const float* go = (const float*)d_in[22], *beo = (const float*)d_in[23];
    const float* mo = (const float*)d_in[24], *vo = (const float*)d_in[25];
    float* out = (float*)d_out;

    const int N = in_sizes[1];  // number of rows / batch ids

    // workspace layout (aligned)
    char* ws = (char*)d_ws;
    _Float16* Wh   = (_Float16*)ws;                                 //  98304 B
    float*    sc   = (float*)(ws + 98304);                          //   1536 B
    float*    tb   = (float*)(ws + 98304 + 1536);                   //   1536 B
    float*    gsum = (float*)(ws + 101376);                         //   1 MB
    unsigned* gmax = (unsigned*)(ws + 101376 + 1048576);            //   1 MB
    int*      gcnt = (int*)(ws + 101376 + 2 * 1048576);             //   8 KB
    (void)ws_size; (void)n_in; (void)out_size;

    // 1) init accumulators + weight conversion + BN folding
    {
        const int total = NSEG * HDIM;  // 262144 covers every sub-task
        prep_kernel<<<(total + 255) / 256, 256, 0, stream>>>(
            W1, W2, W3,
            b1, g1, be1, m1, v1,
            b2, g2, be2, m2, v2,
            b3, g3, be3, m3, v3,
            Wh, sc, tb, gsum, gmax, gcnt);
    }

    // 2) fused MLP + segmented aggregation
    {
        const int ntiles = (N + TILE_ROWS - 1) / TILE_ROWS;
        mlp_agg_kernel<<<ntiles, THREADS, 0, stream>>>(
            x, batch, Wh, sc, tb, gsum, gmax, gcnt, N);
    }

    // 3) concat + output linear + BN
    out_kernel<<<NSEG, ODIM, 0, stream>>>(
        gsum, gmax, gcnt, Wo, bo, go, beo, mo, vo, out);
}
